// AttentionBlock_37151467110517
// MI455X (gfx1250) — matrix operand order repaired and checked by reading, not compiled, not run on hardware
//
#include <hip/hip_runtime.h>

#define BATCH 4
#define NPIX  4096   // H*W = 64*64
#define CCH   256    // channels
#define KCHUNK 32    // keys per attention inner chunk

typedef __attribute__((ext_vector_type(16))) __bf16 v16bf;
typedef __attribute__((ext_vector_type(8)))  float  v8f;
typedef int v4i_t __attribute__((vector_size(16)));   // matches builtin pointee

union FragU { v16bf v; unsigned int u[8]; };

static __device__ inline unsigned short f32_to_bf16(float f) {
    unsigned int u = __float_as_uint(f);
    unsigned int lsb = (u >> 16) & 1u;
    u += 0x7fffu + lsb;                 // round-to-nearest-even
    return (unsigned short)(u >> 16);
}

// ---------------------------------------------------------------------------
// Async global -> LDS copy of one 16-byte segment per lane (ASYNCcnt path).
// Builtin prototype (from probe diagnostic): first param is
//   v4i __attribute__((address_space(1)))*  (global src), then LDS dst.
// ---------------------------------------------------------------------------
#if __has_builtin(__builtin_amdgcn_global_load_async_to_lds_b128)
#define HAVE_ASYNC_B128 1
#endif

static __device__ inline void async_cp16(const unsigned short* g, unsigned short* l) {
#ifdef HAVE_ASYNC_B128
    __builtin_amdgcn_global_load_async_to_lds_b128(
        (__attribute__((address_space(1))) v4i_t*)g,
        (__attribute__((address_space(3))) v4i_t*)l, 0, 0);
#else
    unsigned int loff = (unsigned int)(unsigned long long)
        (__attribute__((address_space(3))) unsigned short*)l;
    asm volatile("global_load_async_to_lds_b128 %0, %1, off"
                 :: "v"(loff), "v"((unsigned long long)(size_t)g)
                 : "memory");
#endif
}

static __device__ inline void wait_async0() {
#if __has_builtin(__builtin_amdgcn_s_wait_asynccnt)
    __builtin_amdgcn_s_wait_asynccnt(0);
#else
    asm volatile("s_wait_asynccnt 0x0" ::: "memory");
#endif
}

// ---------------------------------------------------------------------------
// A-fragment: 16x32 bf16, source row-major [m][k] with row stride ld (elems).
// ISA layout: VGPR v holds packed K pair  k = 16*(v>=4) + 8*(lane>=16) + 2*(v&3).
// ---------------------------------------------------------------------------
static __device__ inline v16bf load_frag_a(const unsigned short* src, int ld,
                                           int row0, int k0, int lane) {
    FragU f;
    int m  = row0 + (lane & 15);
    int kh = (lane >> 4) << 3;                       // +8 for lanes 16..31
    const unsigned short* base = src + (size_t)m * ld + k0 + kh;
#pragma unroll
    for (int v = 0; v < 8; ++v) {
        int k = ((v >> 2) << 4) + ((v & 3) << 1);
        f.u[v] = *(const unsigned int*)(base + k);
    }
    return f.v;
}

// ---------------------------------------------------------------------------
// B-fragment: 32x16 bf16, source stored so element (k,n) sits at src[n*ld + k]
// (k contiguous). ISA layout: lane n = lane&15, VGPR v holds pair
// k = 16*(v>=4) + 8*(lane>=16) + 2*(v&3).
// ---------------------------------------------------------------------------
static __device__ inline v16bf load_frag_b(const unsigned short* src, int ld,
                                           int n0, int k0, int lane) {
    FragU f;
    int n  = n0 + (lane & 15);
    int kh = (lane >> 4) << 3;                       // +8 for lanes 16..31
    const unsigned short* base = src + (size_t)n * ld + k0 + kh;
#pragma unroll
    for (int v = 0; v < 8; ++v)
        f.u[v] = *(const unsigned int*)(base + ((v >> 2) << 4) + ((v & 3) << 1));
    return f.v;
}

static __device__ inline v8f wmma_bf16(v16bf a, v16bf b, v8f c) {
    return __builtin_amdgcn_wmma_f32_16x16x32_bf16(false, a, false, b,
                                                   (short)0, c, false, false);
}

// ---------------------------------------------------------------------------
// Prep kernels
// ---------------------------------------------------------------------------
__global__ void cvt_bf16_kernel(const float* __restrict__ src,
                                unsigned short* __restrict__ dst, int n) {
    int i = blockIdx.x * blockDim.x + threadIdx.x;
    if (i < n) dst[i] = f32_to_bf16(src[i]);
}

// Wt[n*C + k] = W[k*C + n]  (bf16)
__global__ void transpose_w_kernel(const float* __restrict__ W,
                                   unsigned short* __restrict__ Wt) {
    int k = threadIdx.x;        // 0..C-1
    int n = blockIdx.x;         // 0..C-1
    Wt[(size_t)n * CCH + k] = f32_to_bf16(W[(size_t)k * CCH + n]);
}

// ---------------------------------------------------------------------------
// Kernel 1: QKV projection.  block = 128 threads (4 waves), wave owns a
// 16-row tile; grid = (NPIX/64, BATCH).
// ---------------------------------------------------------------------------
__device__ inline void proj_tile(const v16bf* a, const unsigned short* Wt,
                                 const float* bias, unsigned short* dst,
                                 int ldst, bool transposed, int row0, int lane) {
    int nl = lane & 15;
    int rb = (lane >> 4) << 3;
#pragma unroll
    for (int ct = 0; ct < 16; ++ct) {
        v8f acc = {};
#pragma unroll
        for (int kt = 0; kt < 8; ++kt) {
            v16bf bf = load_frag_b(Wt, CCH, ct * 16, kt * 32, lane);
            acc = wmma_bf16(a[kt], bf, acc);
        }
        int n = ct * 16 + nl;
        float bv = bias[n];
#pragma unroll
        for (int r = 0; r < 8; ++r) {
            float val = acc[r] + bv;
            int row = row0 + rb + r;
            if (transposed)
                dst[(size_t)n * ldst + row] = f32_to_bf16(val);   // Vt[c][pix]
            else
                dst[(size_t)row * ldst + n] = f32_to_bf16(val);   // [pix][c]
        }
    }
}

__global__ __launch_bounds__(128) void qkv_kernel(
    const unsigned short* __restrict__ xbf,
    const unsigned short* __restrict__ Wqt,
    const unsigned short* __restrict__ Wkt,
    const unsigned short* __restrict__ Wvt,
    const float* __restrict__ bq, const float* __restrict__ bk,
    const float* __restrict__ bv,
    unsigned short* __restrict__ Qbf, unsigned short* __restrict__ Kbf,
    unsigned short* __restrict__ Vt) {
    int lane = threadIdx.x & 31;
    int wave = threadIdx.x >> 5;
    int b    = blockIdx.y;
    int row0 = (blockIdx.x * 4 + wave) * 16;

    const unsigned short* xb = xbf + (size_t)b * NPIX * CCH;
    v16bf a[8];
#pragma unroll
    for (int kt = 0; kt < 8; ++kt)
        a[kt] = load_frag_a(xb, CCH, row0, kt * 32, lane);

    proj_tile(a, Wqt, bq, Qbf + (size_t)b * NPIX * CCH, CCH, false, row0, lane);
    proj_tile(a, Wkt, bk, Kbf + (size_t)b * NPIX * CCH, CCH, false, row0, lane);
    proj_tile(a, Wvt, bv, Vt  + (size_t)b * CCH * NPIX, NPIX, true, row0, lane);
}

// ---------------------------------------------------------------------------
// Kernel 2: flash attention.  block = 256 threads (8 waves), wave owns 16
// queries (128 queries/block).  K/V chunks staged in LDS via async copies,
// double-buffered; all 8 waves share each staged chunk.
// grid = (NPIX/128, BATCH).
// ---------------------------------------------------------------------------
__global__ __launch_bounds__(256) void attn_kernel(
    const unsigned short* __restrict__ Qbf,
    const unsigned short* __restrict__ Kbf,
    const unsigned short* __restrict__ Vt,
    unsigned short* __restrict__ Abf) {
    __shared__ unsigned short Kl[2][KCHUNK * CCH];    // [key][c]  2 x 16KB
    __shared__ unsigned short Vl[2][CCH * KCHUNK];    // [c][key]  2 x 16KB
    __shared__ unsigned short Plds[8][16 * 32];       // P repack, 8KB

    int lane = threadIdx.x & 31;
    int wave = threadIdx.x >> 5;
    int tid  = threadIdx.x;
    int b    = blockIdx.y;
    int q0   = (blockIdx.x * 8 + wave) * 16;

    const unsigned short* Qb = Qbf + (size_t)b * NPIX * CCH;
    const unsigned short* Kb = Kbf + (size_t)b * NPIX * CCH;
    const unsigned short* Vb = Vt  + (size_t)b * CCH * NPIX;

    v16bf qa[8];
#pragma unroll
    for (int kt = 0; kt < 8; ++kt)
        qa[kt] = load_frag_a(Qb, CCH, q0, kt * 32, lane);

    v8f o[16];
#pragma unroll
    for (int ct = 0; ct < 16; ++ct) o[ct] = (v8f){};
    float m_i[8], l_i[8];
#pragma unroll
    for (int r = 0; r < 8; ++r) { m_i[r] = -3.0e38f; l_i[r] = 0.0f; }

    int nl = lane & 15;
    int rb = (lane >> 4) << 3;
    unsigned short* P = Plds[wave];

    // ---- async stage of one 32-key chunk: K 16KB + V 16KB, 16B/lane x4 ---
    auto issue_chunk = [&](int buf, int j0) {
        const unsigned short* gk = Kb + (size_t)j0 * CCH;   // contiguous 16KB
        unsigned short* lk = Kl[buf];
#pragma unroll
        for (int i = 0; i < 4; ++i) {
            int s = tid + i * 256;                          // 16B segment id
            async_cp16(gk + s * 8, lk + s * 8);
        }
        unsigned short* lv = Vl[buf];
#pragma unroll
        for (int i = 0; i < 4; ++i) {
            int s = tid + i * 256;
            int c = s >> 2, part = s & 3;                   // 4 segs per c-row
            async_cp16(Vb + (size_t)c * NPIX + j0 + part * 8,
                       lv + c * KCHUNK + part * 8);
        }
    };

    const int NITER = NPIX / KCHUNK;
    issue_chunk(0, 0);
    wait_async0();
    __syncthreads();

    int cur = 0;
    for (int it = 0; it < NITER; ++it) {
        int j0 = it * KCHUNK;
        if (it + 1 < NITER) issue_chunk(cur ^ 1, j0 + KCHUNK);

        // ---- scores: two 16x16 tiles over staged keys ------------------
        v8f s0 = {}, s1 = {};
        const unsigned short* kl = Kl[cur];
#pragma unroll
        for (int kt = 0; kt < 8; ++kt) {
            v16bf kf0 = load_frag_b(kl, CCH, 0,  kt * 32, lane);
            v16bf kf1 = load_frag_b(kl, CCH, 16, kt * 32, lane);
            s0 = wmma_bf16(qa[kt], kf0, s0);
            s1 = wmma_bf16(qa[kt], kf1, s1);
        }

        // ---- online softmax (row groups = 16-lane halves) --------------
        float alpha[8];
#pragma unroll
        for (int r = 0; r < 8; ++r) {
            float mx = fmaxf(s0[r], s1[r]);
#pragma unroll
            for (int off = 8; off >= 1; off >>= 1)
                mx = fmaxf(mx, __shfl_xor(mx, off, 32));
            float mnew = fmaxf(m_i[r], mx);
            alpha[r] = __expf(m_i[r] - mnew);
            m_i[r] = mnew;
            float p0 = __expf(s0[r] - mnew);
            float p1 = __expf(s1[r] - mnew);
            s0[r] = p0; s1[r] = p1;
            float rs = p0 + p1;
#pragma unroll
            for (int off = 8; off >= 1; off >>= 1)
                rs += __shfl_xor(rs, off, 32);
            l_i[r] = l_i[r] * alpha[r] + rs;
        }
#pragma unroll
        for (int ct = 0; ct < 16; ++ct)
#pragma unroll
            for (int r = 0; r < 8; ++r) o[ct][r] *= alpha[r];

        // ---- repack P: D-layout (regs) -> A-layout via LDS -------------
        __syncthreads();
#pragma unroll
        for (int r = 0; r < 8; ++r) {
            P[(rb + r) * 32 + nl]      = f32_to_bf16(s0[r]);
            P[(rb + r) * 32 + 16 + nl] = f32_to_bf16(s1[r]);
        }
        __syncthreads();
        v16bf pf = load_frag_a(P, 32, 0, 0, lane);

        // ---- O += P @ V from staged V ----------------------------------
        const unsigned short* vl = Vl[cur];
#pragma unroll
        for (int ct = 0; ct < 16; ++ct) {
            v16bf vf = load_frag_b(vl, KCHUNK, ct * 16, 0, lane);
            o[ct] = wmma_bf16(pf, vf, o[ct]);
        }

        wait_async0();          // next chunk's copies landed (this wave)
        __syncthreads();        // ... and everyone's copies visible
        cur ^= 1;
    }

    // ---- normalize + store attended (bf16 row-major) --------------------
    float inv[8];
#pragma unroll
    for (int r = 0; r < 8; ++r) inv[r] = 1.0f / l_i[r];
    unsigned short* Ab = Abf + (size_t)b * NPIX * CCH;
#pragma unroll
    for (int ct = 0; ct < 16; ++ct)
#pragma unroll
        for (int r = 0; r < 8; ++r)
            Ab[(size_t)(q0 + rb + r) * CCH + ct * 16 + nl] =
                f32_to_bf16(o[ct][r] * inv[r]);
}

// ---------------------------------------------------------------------------
// Kernel 3: output projection + bias + residual  y = x + gamma*(A@Wo + bo)
// ---------------------------------------------------------------------------
__global__ __launch_bounds__(128) void outproj_kernel(
    const unsigned short* __restrict__ Abf,
    const unsigned short* __restrict__ Wot,
    const float* __restrict__ bo,
    const float* __restrict__ x,
    const float* __restrict__ gamma,
    float* __restrict__ y) {
    int lane = threadIdx.x & 31;
    int wave = threadIdx.x >> 5;
    int b    = blockIdx.y;
    int row0 = (blockIdx.x * 4 + wave) * 16;

    const unsigned short* Ab = Abf + (size_t)b * NPIX * CCH;
    v16bf a[8];
#pragma unroll
    for (int kt = 0; kt < 8; ++kt)
        a[kt] = load_frag_a(Ab, CCH, row0, kt * 32, lane);

    float g  = gamma[0];
    int   nl = lane & 15;
    int   rbse = (lane >> 4) << 3;
#pragma unroll
    for (int ct = 0; ct < 16; ++ct) {
        v8f acc = {};
#pragma unroll
        for (int kt = 0; kt < 8; ++kt) {
            v16bf bf = load_frag_b(Wot, CCH, ct * 16, kt * 32, lane);
            acc = wmma_bf16(a[kt], bf, acc);
        }
        int n = ct * 16 + nl;
        float bias = bo[n];
#pragma unroll
        for (int r = 0; r < 8; ++r) {
            size_t idx = ((size_t)b * NPIX + row0 + rbse + r) * CCH + n;
            y[idx] = x[idx] + g * (acc[r] + bias);
        }
    }
}

// ---------------------------------------------------------------------------
// Host launcher
// ---------------------------------------------------------------------------
extern "C" void kernel_launch(void* const* d_in, const int* in_sizes, int n_in,
                              void* d_out, int out_size, void* d_ws, size_t ws_size,
                              hipStream_t stream) {
    (void)in_sizes; (void)n_in; (void)out_size; (void)ws_size;
    const float* x     = (const float*)d_in[0];
    const float* Wq    = (const float*)d_in[1];
    const float* bq    = (const float*)d_in[2];
    const float* Wk    = (const float*)d_in[3];
    const float* bk    = (const float*)d_in[4];
    const float* Wv    = (const float*)d_in[5];
    const float* bv    = (const float*)d_in[6];
    const float* Wo    = (const float*)d_in[7];
    const float* bo    = (const float*)d_in[8];
    const float* gamma = (const float*)d_in[9];
    float* y = (float*)d_out;

    const size_t tensElems = (size_t)BATCH * NPIX * CCH;   // 4.19M
    unsigned short* w   = (unsigned short*)d_ws;
    size_t off = 0;
    unsigned short* xbf = w + off; off += tensElems;
    unsigned short* Wqt = w + off; off += (size_t)CCH * CCH;
    unsigned short* Wkt = w + off; off += (size_t)CCH * CCH;
    unsigned short* Wvt = w + off; off += (size_t)CCH * CCH;
    unsigned short* Wot = w + off; off += (size_t)CCH * CCH;
    unsigned short* Qbf = w + off; off += tensElems;
    unsigned short* Kbf = w + off; off += tensElems;
    unsigned short* Vt  = w + off; off += tensElems;
    unsigned short* Abf = w + off; off += tensElems;

    int total = (int)tensElems;
    cvt_bf16_kernel<<<(total + 255) / 256, 256, 0, stream>>>(x, xbf, total);
    transpose_w_kernel<<<CCH, CCH, 0, stream>>>(Wq, Wqt);
    transpose_w_kernel<<<CCH, CCH, 0, stream>>>(Wk, Wkt);
    transpose_w_kernel<<<CCH, CCH, 0, stream>>>(Wv, Wvt);
    transpose_w_kernel<<<CCH, CCH, 0, stream>>>(Wo, Wot);

    dim3 gridP(NPIX / 64, BATCH);
    qkv_kernel<<<gridP, 128, 0, stream>>>(xbf, Wqt, Wkt, Wvt, bq, bk, bv,
                                          Qbf, Kbf, Vt);
    dim3 gridA(NPIX / 128, BATCH);
    attn_kernel<<<gridA, 256, 0, stream>>>(Qbf, Kbf, Vt, Abf);
    outproj_kernel<<<gridP, 128, 0, stream>>>(Abf, Wot, bo, x, gamma, y);
}
